// RandlaNet_53154515256146
// MI455X (gfx1250) — matrix operand order repaired and checked by reading, not compiled, not run on hardware
//
#include <hip/hip_runtime.h>
#include <hip/hip_bf16.h>
#include <math.h>

// ===========================================================================
// RandLA-Net forward for MI455X (gfx1250, wave32).
//
// Activations are channel-major f32 tensors (B, C, P): [(b*C + c)*P + p].
// Every 1x1 conv (and the attention-score matmul) is a GEMM
//   Y = W(Cout,Cin) * X(Cin,P)
// executed with V_WMMA_F32_16X16X32_F16: f16 fragments converted on load,
// f32 accumulation, bias + eval-mode BN + LeakyReLU fused in the epilogue.
// One wave computes a 16(co) x 64(p) tile: the A (weight) fragment is reused
// across 4 WMMAs.  Main K-loop is completely branch-free (clamped addresses,
// unconditional loads); the ragged-K tail masks values with selects, not
// branches, so loads clause up and latency is covered by one wait.
// The network is HBM-bound (~15 GFLOP vs hundreds of MB of (B,C,N,K)
// tensors at 23.3 TB/s), so conv outputs are written directly into channel
// slices of concat buffers and every tensor is touched once per stage.
// ===========================================================================

#define BATCH 4
#define KNN 16
#define PTILES 4   // 16-point tiles per wave

typedef __attribute__((ext_vector_type(16))) _Float16 v16h;
typedef __attribute__((ext_vector_type(8)))  float    v8f;

// ---------------------------------------------------------------------------
// Fused conv1x1 (+bias +BN +LeakyReLU) via WMMA.
//   Y[b, coOff+co, p] = leaky( bn( sum_ci W[co,ci] * X[b, ciOff+ci, p] + bias ) )
// P must be a multiple of 64.
// ---------------------------------------------------------------------------
__global__ void __launch_bounds__(32)
conv1x1_wmma(const float* __restrict__ X, int XC, int ciOff,
             const float* __restrict__ W, const float* __restrict__ bias,
             const float* __restrict__ bnG, const float* __restrict__ bnB,
             const float* __restrict__ bnM, const float* __restrict__ bnV,
             float eps, float slope, int useBn,
             float* __restrict__ Y, int YC, int coOff,
             int Cin, int Cout, int P)
{
    const int lane = threadIdx.x;
    const int lm   = lane & 15;
    const int h    = lane >> 4;
    const int tco  = blockIdx.x * 16;
    const int tp   = blockIdx.y * (16 * PTILES);
    const int b    = blockIdx.z;

    v8f acc[PTILES];
    #pragma unroll
    for (int t = 0; t < PTILES; ++t) acc[t] = (v8f){};

    // A-matrix row = output channel; rows >= Cout clamp to a valid row and
    // compute garbage that is never stored (no guard needed in main loop).
    int arow = tco + lm;
    if (arow >= Cout) arow = Cout - 1;
    const float* __restrict__ wrow  = W + (size_t)arow * Cin;
    const float* __restrict__ xbase = X + ((size_t)b * XC + ciOff) * P;

    int pcol[PTILES];
    #pragma unroll
    for (int t = 0; t < PTILES; ++t) pcol[t] = tp + 16 * t + lm;

    int k0 = 0;
    // ---- main loop: full 32-wide K blocks, zero guards, zero branches ----
    for (; k0 + 32 <= Cin; k0 += 32) {
        v16h a;
        // A fragment in this code's contiguous k order (B matches): lane = M; VGPR j -> K=2j+16h
        #pragma unroll
        for (int j = 0; j < 8; ++j) {
            int kb = 2 * j + 16 * h;
            a[2 * j]     = (_Float16)wrow[k0 + kb];
            a[2 * j + 1] = (_Float16)wrow[k0 + kb + 1];
        }
        #pragma unroll
        for (int t = 0; t < PTILES; ++t) {
            v16h bm;
            const float* __restrict__ xc = xbase + (size_t)k0 * P + pcol[t];
            // B fragment, same contiguous k order as A: lane = N column; element e -> K = 16h + e
            #pragma unroll
            for (int e = 0; e < 16; ++e)
                bm[e] = (_Float16)xc[(size_t)(h * 16 + e) * P];
            acc[t] = __builtin_amdgcn_wmma_f32_16x16x32_f16(
                false, a, false, bm, (short)0, acc[t], false, false);
        }
    }
    // ---- ragged-K tail (at most once): clamped addresses + value selects ----
    if (k0 < Cin) {
        v16h a;
        #pragma unroll
        for (int j = 0; j < 8; ++j) {
            int kb = 2 * j + 16 * h;
            int k  = k0 + kb;
            float x0 = wrow[(k     < Cin) ? k     : 0];
            float x1 = wrow[(k + 1 < Cin) ? k + 1 : 0];
            a[2 * j]     = (_Float16)((k     < Cin) ? x0 : 0.0f);
            a[2 * j + 1] = (_Float16)((k + 1 < Cin) ? x1 : 0.0f);
        }
        #pragma unroll
        for (int t = 0; t < PTILES; ++t) {
            v16h bm;
            const float* __restrict__ xc = xbase + pcol[t];
            #pragma unroll
            for (int e = 0; e < 16; ++e) {
                int ci = k0 + h * 16 + e;
                float xv = xc[(size_t)((ci < Cin) ? ci : 0) * P];
                bm[e] = (_Float16)((ci < Cin) ? xv : 0.0f);
            }
            acc[t] = __builtin_amdgcn_wmma_f32_16x16x32_f16(
                false, a, false, bm, (short)0, acc[t], false, false);
        }
    }

    // ---- epilogue: fold bias+BN into scale/shift, guard only the store ----
    // C/D layout: lane = N column; VGPR r -> M = r + 8h
    #pragma unroll
    for (int r = 0; r < 8; ++r) {
        int co  = tco + r + 8 * h;
        int cos = (co < Cout) ? co : Cout - 1;
        float shift = bias ? bias[cos] : 0.0f;
        float scale = 1.0f;
        if (useBn) {
            float inv = rsqrtf(bnV[cos] + eps);
            scale = bnG[cos] * inv;
            shift = (shift - bnM[cos]) * scale + bnB[cos];
        }
        if (co < Cout) {
            float* __restrict__ yrow = Y + ((size_t)b * YC + coOff + co) * P;
            #pragma unroll
            for (int t = 0; t < PTILES; ++t) {
                float y = acc[t][r] * scale + shift;
                y = (y >= 0.0f) ? y : slope * y;
                yrow[pcol[t]] = y;
            }
        }
    }
}

// ---------------------------------------------------------------------------
// fc1: features (B,N,10) point-major -> Y (B,8,N) channel-major, + BN + leaky
// ---------------------------------------------------------------------------
__global__ void fc1_bn_leaky(const float* __restrict__ feat,
                             const float* __restrict__ w,   // (8,10)
                             const float* __restrict__ bias,
                             const float* __restrict__ bnB, const float* __restrict__ bnG,
                             const float* __restrict__ bnM, const float* __restrict__ bnV,
                             float* __restrict__ Y, int N)
{
    size_t t = (size_t)blockIdx.x * blockDim.x + threadIdx.x;
    size_t tot = (size_t)BATCH * N;
    if (t >= tot) return;
    int n = (int)(t % N);
    int b = (int)(t / N);
    const float* f = feat + t * 10;
    float xv[10];
    #pragma unroll
    for (int i = 0; i < 10; ++i) xv[i] = f[i];
    #pragma unroll
    for (int co = 0; co < 8; ++co) {
        float s = bias[co];
        #pragma unroll
        for (int ci = 0; ci < 10; ++ci) s += w[co * 10 + ci] * xv[ci];
        float inv = rsqrtf(bnV[co] + 1e-5f);
        s = bnG[co] * (s - bnM[co]) * inv + bnB[co];
        s = (s >= 0.0f) ? s : 0.01f * s;
        Y[((size_t)b * 8 + co) * N + n] = s;
    }
}

// ---------------------------------------------------------------------------
// rel_pos: xyz (B,N,3), idx (B,N,K) -> Y (B,10,N,K) = [dist, rel, cen, nb]
// ---------------------------------------------------------------------------
__global__ void rel_pos_k(const float* __restrict__ xyz, const int* __restrict__ idx,
                          int N, float* __restrict__ Y)
{
    size_t t = (size_t)blockIdx.x * blockDim.x + threadIdx.x;
    size_t tot = (size_t)BATCH * N * KNN;
    if (t >= tot) return;
    int k = (int)(t % KNN); size_t r = t / KNN;
    int n = (int)(r % N);   int b = (int)(r / N);
    const float* cen = xyz + ((size_t)b * N + n) * 3;
    int j = idx[((size_t)b * N + n) * KNN + k];
    const float* nb = xyz + ((size_t)b * N + j) * 3;
    float cx = cen[0], cy = cen[1], cz = cen[2];
    float nx = nb[0],  ny = nb[1],  nz = nb[2];
    float rx = cx - nx, ry = cy - ny, rz = cz - nz;
    float dist = sqrtf(rx * rx + ry * ry + rz * rz + 1e-9f);
    size_t P = (size_t)N * KNN;
    size_t base = (size_t)b * 10 * P + (size_t)n * KNN + k;
    Y[base + 0 * P] = dist;
    Y[base + 1 * P] = rx; Y[base + 2 * P] = ry; Y[base + 3 * P] = rz;
    Y[base + 4 * P] = cx; Y[base + 5 * P] = cy; Y[base + 6 * P] = cz;
    Y[base + 7 * P] = nx; Y[base + 8 * P] = ny; Y[base + 9 * P] = nz;
}

// ---------------------------------------------------------------------------
// gather_neighbour into a channel slice: feat (B,C,N), idx (B,M,K)
//   Y[b, cOff+c, m, k] = feat[b, c, idx[b,m,k]]        (Y has YC channels)
// ---------------------------------------------------------------------------
__global__ void gather_nb(const float* __restrict__ feat, int C, int N,
                          const int* __restrict__ idx, int M,
                          float* __restrict__ Y, int YC, int cOff)
{
    size_t t = (size_t)blockIdx.x * blockDim.x + threadIdx.x;
    size_t tot = (size_t)BATCH * C * M * KNN;
    if (t >= tot) return;
    int k = (int)(t % KNN); size_t r = t / KNN;
    int m = (int)(r % M); r /= M;
    int c = (int)(r % C); int b = (int)(r / C);
    int j = idx[((size_t)b * M + m) * KNN + k];
    Y[(((size_t)b * YC + cOff + c) * M + m) * KNN + k] =
        feat[((size_t)b * C + c) * N + j];
}

// ---------------------------------------------------------------------------
// attention pool core: x,s (B,C,N,K); per (b,c,n) softmax s over K, weighted sum
// ---------------------------------------------------------------------------
__global__ void att_softmax_agg(const float* __restrict__ x, const float* __restrict__ s,
                                int C, int N, float* __restrict__ Y)
{
    size_t t = (size_t)blockIdx.x * blockDim.x + threadIdx.x;
    size_t tot = (size_t)BATCH * C * N;
    if (t >= tot) return;
    int n = (int)(t % N); size_t r = t / N;
    int c = (int)(r % C); int b = (int)(r / C);
    size_t base = (((size_t)b * C + c) * N + n) * KNN;
    float mx = -3.4e38f;
    #pragma unroll
    for (int k = 0; k < KNN; ++k) mx = fmaxf(mx, s[base + k]);
    float se = 0.0f, acc = 0.0f;
    #pragma unroll
    for (int k = 0; k < KNN; ++k) {
        float e = __expf(s[base + k] - mx);
        se += e;
        acc += e * x[base + k];
    }
    Y[((size_t)b * C + c) * N + n] = acc / se;
}

// ---------------------------------------------------------------------------
// random_sample: feat (B,C,N), sub (B,M,K) -> Y (B,C,M) = max over K gathered
// ---------------------------------------------------------------------------
__global__ void max_pool_gather(const float* __restrict__ feat, int C, int N,
                                const int* __restrict__ sub, int M,
                                float* __restrict__ Y)
{
    size_t t = (size_t)blockIdx.x * blockDim.x + threadIdx.x;
    size_t tot = (size_t)BATCH * C * M;
    if (t >= tot) return;
    int m = (int)(t % M); size_t r = t / M;
    int c = (int)(r % C); int b = (int)(r / C);
    float mx = -3.4e38f;
    #pragma unroll
    for (int k = 0; k < KNN; ++k) {
        int j = sub[((size_t)b * M + m) * KNN + k];
        mx = fmaxf(mx, feat[((size_t)b * C + c) * N + j]);
    }
    Y[((size_t)b * C + c) * M + m] = mx;
}

// ---------------------------------------------------------------------------
// interp into a channel slice: feat (B,C,Nc), idx (B,Nf,1)
// ---------------------------------------------------------------------------
__global__ void interp_slice(const float* __restrict__ feat, int C, int Nc,
                             const int* __restrict__ idx, int Nf,
                             float* __restrict__ Y, int YC, int cOff)
{
    size_t t = (size_t)blockIdx.x * blockDim.x + threadIdx.x;
    size_t tot = (size_t)BATCH * C * Nf;
    if (t >= tot) return;
    int n = (int)(t % Nf); size_t r = t / Nf;
    int c = (int)(r % C); int b = (int)(r / C);
    int j = idx[(size_t)b * Nf + n];
    Y[((size_t)b * YC + cOff + c) * Nf + n] = feat[((size_t)b * C + c) * Nc + j];
}

// ---------------------------------------------------------------------------
// copy (B,C,N) into channel slice of (B,YC,N)
// ---------------------------------------------------------------------------
__global__ void copy_slice(const float* __restrict__ X, int C, int N,
                           float* __restrict__ Y, int YC, int cOff)
{
    size_t t = (size_t)blockIdx.x * blockDim.x + threadIdx.x;
    size_t tot = (size_t)BATCH * C * N;
    if (t >= tot) return;
    int n = (int)(t % N); size_t r = t / N;
    int c = (int)(r % C); int b = (int)(r / C);
    Y[((size_t)b * YC + cOff + c) * N + n] = X[((size_t)b * C + c) * N + n];
}

// residual: Y = leaky(Aa + Bb, slope)
__global__ void add_leaky(const float* __restrict__ Aa, const float* __restrict__ Bb,
                          float slope, float* __restrict__ Y, size_t n)
{
    size_t t = (size_t)blockIdx.x * blockDim.x + threadIdx.x;
    if (t >= n) return;
    float v = Aa[t] + Bb[t];
    Y[t] = (v >= 0.0f) ? v : slope * v;
}

// final: X (B,C,N) -> out (B*N, C)
__global__ void transpose_out(const float* __restrict__ X, int C, int N,
                              float* __restrict__ out)
{
    size_t t = (size_t)blockIdx.x * blockDim.x + threadIdx.x;
    size_t tot = (size_t)BATCH * C * N;
    if (t >= tot) return;
    int n = (int)(t % N); size_t r = t / N;
    int c = (int)(r % C); int b = (int)(r / C);
    out[((size_t)b * N + n) * C + c] = X[((size_t)b * C + c) * N + n];
}

// ===========================================================================
// Host orchestration
// ===========================================================================
static inline void gemm(hipStream_t st, const float* X, int XC, int ciOff,
                        const float* W, const float* bias,
                        const float* g, const float* bb, const float* m, const float* v,
                        float eps, float slope, int useBn,
                        float* Y, int YC, int coOff, int Cin, int Cout, int P)
{
    dim3 grid((unsigned)((Cout + 15) / 16), (unsigned)(P / (16 * PTILES)), BATCH);
    conv1x1_wmma<<<grid, 32, 0, st>>>(X, XC, ciOff, W, bias, g, bb, m, v,
                                      eps, slope, useBn, Y, YC, coOff, Cin, Cout, P);
}

static inline unsigned blocksFor(size_t total) { return (unsigned)((total + 255) / 256); }

extern "C" void kernel_launch(void* const* d_in, const int* in_sizes, int n_in,
                              void* d_out, int out_size, void* d_ws, size_t ws_size,
                              hipStream_t stream)
{
    (void)in_sizes; (void)n_in; (void)out_size; (void)ws_size;
#define FP(i) ((const float*)d_in[(i)])
#define IP(i) ((const int*)d_in[(i)])

    // -------- input map (setup_inputs dict order) --------
    const float* features = FP(0);
    const float* xyz[4]; const int* neigh[4]; const int* sub[4]; const int* itp[4];
    for (int i = 0; i < 4; ++i) {
        xyz[i]   = FP(1 + 4 * i);
        neigh[i] = IP(2 + 4 * i);
        sub[i]   = IP(3 + 4 * i);
        itp[i]   = IP(4 + 4 * i);
    }
    const int PB = 17;  // params pytree leaves start (sorted-dict-key order)
    // PB+0..3  bn1{b,g,m,v}   PB+4..7 bn2  PB+8..11 bn3  PB+12..15 bn4
    // PB+16 conv2.b PB+17 conv2.w  PB+18/19 conv3  PB+20/21 conv4  PB+22/23 conv5
    // PB+24+6i: dec[i]{bn.b,bn.g,bn.m,bn.v,conv.b,conv.w}
    // PB+48+44i: enc[i]{ap1(bn b,g,m,v; conv.b; conv.w; fc), ap2(...), bb_bn1(4),
    //                   bb_bn2(4), bb_conv1(b,w), bb_conv2(b,w), bn1(4), bn2(4),
    //                   bn_sc(4), conv1(b,w), conv2(b,w), sc(b,w)}
    // PB+224 fc1_b   PB+225 fc1_w

    // -------- workspace bump allocation (floats) --------
    float* wsf = (float*)d_ws;
    size_t off = 0;
    auto WS = [&](size_t nElem) {
        float* p = wsf + off;
        off += (nElem + 63) & ~(size_t)63;
        return p;
    };
    float* e0 = WS((size_t)BATCH * 32 * 16384);   // encoded[0]
    float* e1 = WS((size_t)BATCH * 32 * 4096);    // encoded[1]
    float* e2 = WS((size_t)BATCH * 128 * 1024);   // encoded[2]
    float* e3 = WS((size_t)BATCH * 256 * 256);    // encoded[3]
    float* e4 = WS((size_t)BATCH * 512 * 64);     // encoded[4]
    const size_t NKA = (size_t)BATCH * 16 * 16384 * KNN;  // 16.8M floats
    float* NK1 = WS(NKA);   // cat1 / decoder concat
    float* NK2 = WS(NKA);   // cat2 / conv3 out
    float* NK3 = WS(NKA);   // rel_pos / attention scores
    const size_t SMA = (size_t)BATCH * 32 * 16384;        // 2.1M floats
    float* SM1 = WS(SMA); float* SM2 = WS(SMA); float* SM3 = WS(SMA);
    float* SM4 = WS(SMA); float* SM5 = WS(SMA); float* SM6 = WS(SMA);
    float* x0  = WS((size_t)BATCH * 8 * 16384);

    const int NSa[5] = {16384, 4096, 1024, 256, 64};
    const int DO[4]  = {16, 64, 128, 256};
    float* encbuf[5] = {e0, e1, e2, e3, e4};

    // -------- fc1 + bn1 + leaky(0.01) --------
    fc1_bn_leaky<<<blocksFor((size_t)BATCH * NSa[0]), 256, 0, stream>>>(
        features, FP(PB + 225), FP(PB + 224),
        FP(PB + 0), FP(PB + 1), FP(PB + 2), FP(PB + 3), x0, NSa[0]);

    // -------- encoder --------
    const float* xin = x0;
    int fin = 8;
    for (int i = 0; i < 4; ++i) {
        const int N = NSa[i], d = DO[i], dh = d / 2;
        const int Pnk = N * KNN;
        const int eb = PB + 48 + 44 * i;
        const float *ap1_bnB = FP(eb+0), *ap1_bnG = FP(eb+1), *ap1_bnM = FP(eb+2), *ap1_bnV = FP(eb+3);
        const float *ap1_cb = FP(eb+4), *ap1_cw = FP(eb+5), *ap1_fc = FP(eb+6);
        const float *ap2_bnB = FP(eb+7), *ap2_bnG = FP(eb+8), *ap2_bnM = FP(eb+9), *ap2_bnV = FP(eb+10);
        const float *ap2_cb = FP(eb+11), *ap2_cw = FP(eb+12), *ap2_fc = FP(eb+13);
        const float *bbb1B = FP(eb+14), *bbb1G = FP(eb+15), *bbb1M = FP(eb+16), *bbb1V = FP(eb+17);
        const float *bbb2B = FP(eb+18), *bbb2G = FP(eb+19), *bbb2M = FP(eb+20), *bbb2V = FP(eb+21);
        const float *bbc1b = FP(eb+22), *bbc1w = FP(eb+23);
        const float *bbc2b = FP(eb+24), *bbc2w = FP(eb+25);
        const float *bn1B = FP(eb+26), *bn1G = FP(eb+27), *bn1M = FP(eb+28), *bn1V = FP(eb+29);
        const float *bn2B = FP(eb+30), *bn2G = FP(eb+31), *bn2M = FP(eb+32), *bn2V = FP(eb+33);
        const float *bnsB = FP(eb+34), *bnsG = FP(eb+35), *bnsM = FP(eb+36), *bnsV = FP(eb+37);
        const float *c1b = FP(eb+38), *c1w = FP(eb+39);
        const float *c2b = FP(eb+40), *c2w = FP(eb+41);
        const float *scb = FP(eb+42), *scw = FP(eb+43);

        // 1) f1 = leaky(bn1(conv1(x)), 0.2)             (B, d/2, N) -> SM1
        gemm(stream, xin, fin, 0, c1w, c1b, bn1G, bn1B, bn1M, bn1V,
             1e-5f, 0.2f, 1, SM1, dh, 0, fin, dh, N);
        // 2) rel_pos -> NK3 (B,10,N,K)
        rel_pos_k<<<blocksFor((size_t)BATCH * N * KNN), 256, 0, stream>>>(
            xyz[i], neigh[i], N, NK3);
        // 3) f_xyz = leaky(bb_bn1(bb_conv1(rel_pos)),0.2) into cat1 ch[dh:d]
        gemm(stream, NK3, 10, 0, bbc1w, bbc1b, bbb1G, bbb1B, bbb1M, bbb1V,
             1e-6f, 0.2f, 1, NK1, d, dh, 10, dh, Pnk);
        // 4) f_nb = gather(f1) into cat1 ch[0:dh]
        gather_nb<<<blocksFor((size_t)BATCH * dh * N * KNN), 256, 0, stream>>>(
            SM1, dh, N, neigh[i], N, NK1, d, 0);
        // 5) scores1 = ap1.fc * cat1 -> NK3 (B,d,N*K)
        gemm(stream, NK1, d, 0, ap1_fc, nullptr, nullptr, nullptr, nullptr, nullptr,
             0.0f, 1.0f, 0, NK3, d, 0, d, d, Pnk);
        // 6) agg1 = softmax-weighted sum -> SM2 (B,d,N)
        att_softmax_agg<<<blocksFor((size_t)BATCH * d * N), 256, 0, stream>>>(
            NK1, NK3, d, N, SM2);
        // 7) a1 = leaky(ap1.bn(ap1.conv(agg1)),0.2) -> SM3 (B,dh,N)
        gemm(stream, SM2, d, 0, ap1_cw, ap1_cb, ap1_bnG, ap1_bnB, ap1_bnM, ap1_bnV,
             1e-6f, 0.2f, 1, SM3, dh, 0, d, dh, N);
        // 8) f_xyz2 = leaky(bb_bn2(bb_conv2(f_xyz)),0.2) into cat2 ch[dh:d]
        gemm(stream, NK1, d, dh, bbc2w, bbc2b, bbb2G, bbb2B, bbb2M, bbb2V,
             1e-6f, 0.2f, 1, NK2, d, dh, dh, dh, Pnk);
        // 9) f_nb2 = gather(a1) into cat2 ch[0:dh]
        gather_nb<<<blocksFor((size_t)BATCH * dh * N * KNN), 256, 0, stream>>>(
            SM3, dh, N, neigh[i], N, NK2, d, 0);
        // 10) scores2 = ap2.fc * cat2 -> NK3
        gemm(stream, NK2, d, 0, ap2_fc, nullptr, nullptr, nullptr, nullptr, nullptr,
             0.0f, 1.0f, 0, NK3, d, 0, d, d, Pnk);
        // 11) agg2 -> SM2 (B,d,N)
        att_softmax_agg<<<blocksFor((size_t)BATCH * d * N), 256, 0, stream>>>(
            NK2, NK3, d, N, SM2);
        // 12) f_bb = leaky(ap2.bn(ap2.conv(agg2)),0.2) -> SM4 (B,d,N)
        gemm(stream, SM2, d, 0, ap2_cw, ap2_cb, ap2_bnG, ap2_bnB, ap2_bnM, ap2_bnV,
             1e-6f, 0.2f, 1, SM4, d, 0, d, d, N);
        // 13) f2 = bn2(conv2(f_bb)) (identity act) -> SM5 (B,2d,N)
        gemm(stream, SM4, d, 0, c2w, c2b, bn2G, bn2B, bn2M, bn2V,
             1e-5f, 1.0f, 1, SM5, 2 * d, 0, d, 2 * d, N);
        // 14) sc = bn_sc(sc(x)) -> SM6 (B,2d,N)
        gemm(stream, xin, fin, 0, scw, scb, bnsG, bnsB, bnsM, bnsV,
             1e-5f, 1.0f, 1, SM6, 2 * d, 0, fin, 2 * d, N);
        // 15) drb = leaky(f2+sc, 0.01)
        float* drb = (i == 0) ? e0 : SM4;
        add_leaky<<<blocksFor((size_t)BATCH * 2 * d * N), 256, 0, stream>>>(
            SM5, SM6, 0.01f, drb, (size_t)BATCH * 2 * d * N);
        // 16) subsample -> encoded[i+1]
        max_pool_gather<<<blocksFor((size_t)BATCH * 2 * d * NSa[i + 1]), 256, 0, stream>>>(
            drb, 2 * d, N, sub[i], NSa[i + 1], encbuf[i + 1]);
        xin = encbuf[i + 1];
        fin = 2 * d;
    }

    // -------- mid conv2 (512->512 @ N=64), bn2, leaky 0.2 --------
    gemm(stream, e4, 512, 0, FP(PB + 17), FP(PB + 16),
         FP(PB + 5), FP(PB + 4), FP(PB + 6), FP(PB + 7),
         1e-5f, 0.2f, 1, SM1, 512, 0, 512, 512, 64);

    // -------- decoder --------
    float* xd = SM1; int curC = 512, curN = 64;
    float* encsk[4]   = {e3, e2, e1, e0};
    const int CeA[4]  = {256, 128, 32, 32};
    const int NfA[4]  = {256, 1024, 4096, 16384};
    const int* itpA[4] = {itp[3], itp[2], itp[1], itp[0]};
    const int foA[4]  = {256, 128, 32, 32};
    float* decOut[4]  = {SM2, SM3, SM2, SM3};
    for (int i = 0; i < 4; ++i) {
        const int Ce = CeA[i], Nf = NfA[i], fo = foA[i];
        const int fiC = Ce + curC;
        // concat buffer NK1 = [encoded_skip | interp(x)]
        copy_slice<<<blocksFor((size_t)BATCH * Ce * Nf), 256, 0, stream>>>(
            encsk[i], Ce, Nf, NK1, fiC, 0);
        interp_slice<<<blocksFor((size_t)BATCH * curC * Nf), 256, 0, stream>>>(
            xd, curC, curN, itpA[i], Nf, NK1, fiC, Ce);
        const int db = PB + 24 + 6 * i;
        gemm(stream, NK1, fiC, 0, FP(db + 5), FP(db + 4),
             FP(db + 1), FP(db + 0), FP(db + 2), FP(db + 3),
             1e-5f, 1.0f, 1, decOut[i], fo, 0, fiC, fo, Nf);
        xd = decOut[i]; curC = fo; curN = Nf;
    }

    // -------- head: conv3 -> conv4 -> conv5 -> transpose --------
    gemm(stream, xd, 32, 0, FP(PB + 19), FP(PB + 18),
         FP(PB + 9), FP(PB + 8), FP(PB + 10), FP(PB + 11),
         1e-5f, 0.2f, 1, NK2, 64, 0, 32, 64, 16384);
    gemm(stream, NK2, 64, 0, FP(PB + 21), FP(PB + 20),
         FP(PB + 13), FP(PB + 12), FP(PB + 14), FP(PB + 15),
         1e-5f, 0.2f, 1, SM5, 32, 0, 64, 32, 16384);
    gemm(stream, SM5, 32, 0, FP(PB + 23), FP(PB + 22),
         nullptr, nullptr, nullptr, nullptr,
         0.0f, 1.0f, 0, SM6, 13, 0, 32, 13, 16384);
    transpose_out<<<blocksFor((size_t)BATCH * 13 * 16384), 256, 0, stream>>>(
        SM6, 13, 16384, (float*)d_out);

#undef FP
#undef IP
}
